// EmNonLocalLayer_10788957847971
// MI455X (gfx1250) — matrix operand order repaired and checked by reading, not compiled, not run on hardware
//
#include <hip/hip_runtime.h>

// Problem constants (from reference)
#define BB   8
#define CIN  512
#define CI   256
#define COUT 512
#define HH   99
#define WWD  99
#define NN   (HH * WWD)   // 9801
#define KBASES 81
#define POOLW 11          // 99/9
#define EPSV 1e-5f

typedef __attribute__((ext_vector_type(16))) __bf16        v16bf;
typedef __attribute__((ext_vector_type(8)))  float         v8f;
typedef __attribute__((ext_vector_type(4)))  unsigned int  v4u;

// GEMM tiling: block = 64(M) x 128(N) x 32(K), 4 waves; each wave: 64M x 32N
#define MT 64
#define NTILE 128
#define KT 32
#define LDSK 40   // padded LDS row stride in bf16 elems (80B = 20 banks/row, conflict-free)
#define AITER ((MT * KT / 2) / 128)     // 8
#define BITER ((NTILE * KT / 2) / 128)  // 16

__device__ __forceinline__ unsigned short f2bf(float f) {
  // round-to-nearest-even float -> bf16 bits
  unsigned int u = __float_as_uint(f);
  u += 0x7FFFu + ((u >> 16) & 1u);
  return (unsigned short)(u >> 16);
}

__device__ __forceinline__ unsigned int pack2(float v0, float v1) {
  return (unsigned int)f2bf(v0) | ((unsigned int)f2bf(v1) << 16);
}

// Generic batched strided GEMM with optional split-K:
//   C[m,n] (+)= scale * sum_k A[m,k]*B[k,n] (+ bias[m])
// A[m*sAm + k*sAk], B[k*sBk + n*sBn] (int strides, per-batch offsets fit in i32).
// splitK>1: partials accumulated via global_atomic_add_f32 (C pre-zeroed).
__global__ __launch_bounds__(128)
void gemm_bf16_wmma(const float* __restrict__ A, const float* __restrict__ Bm,
                    float* __restrict__ C, const float* __restrict__ bias,
                    int M, int Ndim, int Kdim,
                    int sAm, int sAk, int sBk, int sBn, int ldc,
                    long batA, long batB, long batC, float scale, int splitK)
{
  __shared__ unsigned short lds[(MT + NTILE) * LDSK];
  unsigned short* la = lds;              // A tile: [64 rows(m)][32 k]
  unsigned short* lb = lds + MT * LDSK;  // B tile: [128 rows(n)][32 k]

  const int tid   = threadIdx.x;
  const int lane  = tid & 31;
  const int wv    = tid >> 5;        // wave 0..3 -> 32-wide N slab
  const int half  = lane >> 4;       // 0/1
  const int l16   = lane & 15;
  const int mBase = blockIdx.y * MT;
  const int nBase = blockIdx.x * NTILE;
  const int bz    = blockIdx.z / splitK;
  const int kz    = blockIdx.z % splitK;

  // split-K range (KT-aligned chunks)
  const int kcPer = ((Kdim + splitK * KT - 1) / (splitK * KT)) * KT;
  const int kBeg  = kz * kcPer;
  const int kEnd  = (kBeg + kcPer < Kdim) ? (kBeg + kcPer) : Kdim;
  if (splitK > 1 && kBeg >= kEnd) return;   // uniform per block

  A  += (size_t)bz * batA;
  Bm += (size_t)bz * batB;
  C  += (size_t)bz * batC;

  // Per-thread fill offsets, hoisted out of the K loop (all i32)
  int aOff[AITER], aLds[AITER];
  #pragma unroll
  for (int it = 0; it < AITER; ++it) {
    int i = tid + it * 128;
    int r = i >> 4, kp = (i & 15) << 1;
    aOff[it] = (mBase + r) * sAm + kp * sAk;
    aLds[it] = r * LDSK + kp;
  }
  int bOff[BITER], bLds[BITER];
  #pragma unroll
  for (int it = 0; it < BITER; ++it) {
    int i = tid + it * 128;
    int r = i >> 4, kp = (i & 15) << 1;
    bOff[it] = (nBase + r) * sBn + kp * sBk;
    bLds[it] = r * LDSK + kp;
  }

  const bool mFull = (mBase + MT <= M);
  const bool nFull = (nBase + NTILE <= Ndim);

  v8f acc[4][2];
  #pragma unroll
  for (int mt = 0; mt < 4; ++mt)
    #pragma unroll
    for (int nf = 0; nf < 2; ++nf)
      #pragma unroll
      for (int r = 0; r < 8; ++r) acc[mt][nf][r] = 0.0f;

  union Frag { v4u u[2]; v16bf v; };

  int kA = kBeg * sAk;
  int kB = kBeg * sBk;
  for (int k0 = kBeg; k0 < kEnd; k0 += KT, kA += KT * sAk, kB += KT * sBk) {
    const bool kFull = (k0 + KT <= Kdim);

    // ---- Stage A tile ----
    if (mFull & kFull) {           // fast: unconditional loads, no clamps
      #pragma unroll
      for (int it = 0; it < AITER; ++it) {
        const float* p = A + (aOff[it] + kA);
        float v0 = p[0];
        float v1 = p[sAk];
        *(unsigned int*)&la[aLds[it]] = pack2(v0, v1);
      }
    } else {                       // rare boundary path: clamp + select
      #pragma unroll
      for (int it = 0; it < AITER; ++it) {
        int i = tid + it * 128;
        int r = i >> 4, kp = (i & 15) << 1;
        int gm = mBase + r, gk = k0 + kp;
        int cm = gm < M ? gm : M - 1;
        int c0 = gk < Kdim ? gk : Kdim - 1;
        int c1 = gk + 1 < Kdim ? gk + 1 : Kdim - 1;
        float v0 = A[cm * sAm + c0 * sAk];
        float v1 = A[cm * sAm + c1 * sAk];
        bool okm = gm < M;
        v0 = (okm && gk     < Kdim) ? v0 : 0.0f;
        v1 = (okm && gk + 1 < Kdim) ? v1 : 0.0f;
        *(unsigned int*)&la[r * LDSK + kp] = pack2(v0, v1);
      }
    }

    // ---- Stage B tile ----
    if (nFull & kFull) {
      #pragma unroll
      for (int it = 0; it < BITER; ++it) {
        const float* p = Bm + (bOff[it] + kB);
        float v0 = p[0];
        float v1 = p[sBk];
        *(unsigned int*)&lb[bLds[it]] = pack2(v0, v1);
      }
    } else {
      #pragma unroll
      for (int it = 0; it < BITER; ++it) {
        int i = tid + it * 128;
        int r = i >> 4, kp = (i & 15) << 1;
        int gn = nBase + r, gk = k0 + kp;
        int cn = gn < Ndim ? gn : Ndim - 1;
        int c0 = gk < Kdim ? gk : Kdim - 1;
        int c1 = gk + 1 < Kdim ? gk + 1 : Kdim - 1;
        float v0 = Bm[c0 * sBk + cn * sBn];
        float v1 = Bm[c1 * sBk + cn * sBn];
        bool okn = gn < Ndim;
        v0 = (okn && gk     < Kdim) ? v0 : 0.0f;
        v1 = (okn && gk + 1 < Kdim) ? v1 : 0.0f;
        *(unsigned int*)&lb[r * LDSK + kp] = pack2(v0, v1);
      }
    }

    // prefetch next K-step of A (global_prefetch_b8)
    if (k0 + KT < kEnd) {
      int gm = mBase + (tid & 63);
      int cm = gm < M ? gm : M - 1;
      __builtin_prefetch(&A[cm * sAm + (k0 + KT) * sAk], 0, 3);
    }
    __syncthreads();

    // ---- Compute: 8 WMMAs per wave per K-step ----
    // B fragments (ISA 7.12.2): lane -> col l%16, 8-K runs at (l/16)*8 and +16
    Frag fb[2];
    #pragma unroll
    for (int nf = 0; nf < 2; ++nf) {
      const unsigned short* p = lb + (wv * 32 + nf * 16 + l16) * LDSK + half * 8;
      fb[nf].u[0] = *(const v4u*)(p);
      fb[nf].u[1] = *(const v4u*)(p + 16);
    }
    // A fragments: lane -> row l%16, 8-K runs at (l/16)*8 and +16
    #pragma unroll
    for (int mt = 0; mt < 4; ++mt) {
      Frag fa;
      const unsigned short* p = la + (mt * 16 + l16) * LDSK + half * 8;
      fa.u[0] = *(const v4u*)(p);
      fa.u[1] = *(const v4u*)(p + 16);
      acc[mt][0] = __builtin_amdgcn_wmma_f32_16x16x32_bf16(
          false, fa.v, false, fb[0].v, (short)0, acc[mt][0], false, false);
      acc[mt][1] = __builtin_amdgcn_wmma_f32_16x16x32_bf16(
          false, fa.v, false, fb[1].v, (short)0, acc[mt][1], false, false);
    }
    __syncthreads();
  }

  // ---- Writeback: D lane -> n = l%16, m = r + 8*(l/16) within 16x16 tile ----
  #pragma unroll
  for (int mt = 0; mt < 4; ++mt) {
    #pragma unroll
    for (int nf = 0; nf < 2; ++nf) {
      #pragma unroll
      for (int r = 0; r < 8; ++r) {
        int m = mBase + mt * 16 + r + half * 8;
        int n = nBase + wv * 32 + nf * 16 + l16;
        if (m < M && n < Ndim) {
          float v = acc[mt][nf][r] * scale;
          if (bias && kz == 0) v += bias[m];
          if (splitK > 1) atomicAdd(&C[m * ldc + n], v);
          else            C[m * ldc + n] = v;
        }
      }
    }
  }
}

__global__ __launch_bounds__(256)
void fill_zero(float* __restrict__ p, int n)
{
  int i = blockIdx.x * 256 + threadIdx.x;
  if (i < n) p[i] = 0.0f;
}

// Adaptive 9x9 avg pool of value [B,CI,99,99] -> pooled [B,CI,81]
__global__ __launch_bounds__(256)
void pool9(const float* __restrict__ value, float* __restrict__ pooled)
{
  int idx = blockIdx.x * 256 + threadIdx.x;
  if (idx >= BB * CI * KBASES) return;
  int pw = idx % 9;
  int ph = (idx / 9) % 9;
  int c  = (idx / KBASES) % CI;
  int b  = idx / (KBASES * CI);
  const float* base = value + ((size_t)b * CI + c) * NN;
  float s = 0.0f;
  for (int i = 0; i < POOLW; ++i)
    for (int j = 0; j < POOLW; ++j)
      s += base[(ph * POOLW + i) * WWD + pw * POOLW + j];
  pooled[idx] = s * (1.0f / (POOLW * POOLW));
}

// In-place row softmax; one block per contiguous row of `len` floats
__global__ __launch_bounds__(256)
void softmax_rows(float* __restrict__ data, int len)
{
  __shared__ float red[256];
  float* p = data + (size_t)blockIdx.x * len;
  int t = threadIdx.x;
  float mx = -3.402823466e+38f;
  for (int i = t; i < len; i += 256) mx = fmaxf(mx, p[i]);
  red[t] = mx; __syncthreads();
  for (int s = 128; s > 0; s >>= 1) {
    if (t < s) red[t] = fmaxf(red[t], red[t + s]);
    __syncthreads();
  }
  mx = red[0]; __syncthreads();
  float sum = 0.0f;
  for (int i = t; i < len; i += 256) { float e = __expf(p[i] - mx); p[i] = e; sum += e; }
  red[t] = sum; __syncthreads();
  for (int s = 128; s > 0; s >>= 1) {
    if (t < s) red[t] += red[t + s];
    __syncthreads();
  }
  float inv = 1.0f / red[0];
  for (int i = t; i < len; i += 256) p[i] *= inv;
}

__global__ __launch_bounds__(256)
void add_inplace(float* __restrict__ a, const float* __restrict__ b, int n)
{
  int i = blockIdx.x * 256 + threadIdx.x;
  if (i < n) a[i] += b[i];
}

// BatchNorm training stats: per channel over (B, N)
__global__ __launch_bounds__(256)
void bn_stats(const float* __restrict__ q, float* __restrict__ mean, float* __restrict__ istd)
{
  __shared__ float s1[256], s2[256];
  int c = blockIdx.x, t = threadIdx.x;
  float a = 0.0f, b2 = 0.0f;
  for (int b = 0; b < BB; ++b) {
    const float* p = q + ((size_t)b * CI + c) * NN;
    for (int i = t; i < NN; i += 256) { float v = p[i]; a += v; b2 += v * v; }
  }
  s1[t] = a; s2[t] = b2; __syncthreads();
  for (int s = 128; s > 0; s >>= 1) {
    if (t < s) { s1[t] += s1[t + s]; s2[t] += s2[t + s]; }
    __syncthreads();
  }
  if (t == 0) {
    const float invN = 1.0f / (float)(BB * NN);
    float m = s1[0] * invN;
    float var = s2[0] * invN - m * m;
    mean[c] = m;
    istd[c] = rsqrtf(var + EPSV);
  }
}

__global__ __launch_bounds__(256)
void bn_apply(float* __restrict__ q, const float* __restrict__ mean,
              const float* __restrict__ istd, const float* __restrict__ gamma,
              const float* __restrict__ beta)
{
  size_t i = (size_t)blockIdx.x * 256 + threadIdx.x;
  if (i >= (size_t)BB * CI * NN) return;
  int c = (int)((i / NN) % CI);
  q[i] = (q[i] - mean[c]) * istd[c] * gamma[c] + beta[c];
}

extern "C" void kernel_launch(void* const* d_in, const int* in_sizes, int n_in,
                              void* d_out, int out_size, void* d_ws, size_t ws_size,
                              hipStream_t stream)
{
  (void)in_sizes; (void)n_in; (void)out_size; (void)ws_size;
  const float* x       = (const float*)d_in[0];   // [B,CIN,99,99]
  const float* w_value = (const float*)d_in[1];   // [CI,CIN]
  const float* b_value = (const float*)d_in[2];
  const float* w_vconv = (const float*)d_in[3];   // [CI,CI]
  const float* b_vconv = (const float*)d_in[4];
  const float* w_query = (const float*)d_in[5];   // [CI,CIN]
  const float* b_query = (const float*)d_in[6];
  const float* gamma   = (const float*)d_in[7];
  const float* beta    = (const float*)d_in[8];
  const float* w_out   = (const float*)d_in[9];   // [COUT,CI]
  const float* b_out   = (const float*)d_in[10];
  float* out = (float*)d_out;                     // [B,COUT,99,99]

  // Workspace layout (floats)
  float* ws = (float*)d_ws;
  size_t o = 0;
  float* value  = ws + o; o += (size_t)BB * CI * NN;      // [B,CI,N]
  float* qraw   = ws + o; o += (size_t)BB * CI * NN;      // [B,CI,N]
  float* ctx    = ws + o; o += (size_t)BB * CI * NN;      // [B,CI,N]
  float* simb   = ws + o; o += (size_t)BB * KBASES * NN;  // [B,81,N]
  float* simmap = ws + o; o += (size_t)BB * NN * KBASES;  // [B,N,81]
  float* pooled = ws + o; o += (size_t)BB * CI * KBASES;  // [B,CI,81]
  float* vh     = ws + o; o += (size_t)BB * CI * KBASES;  // [B,CI,81]
  float* bases  = ws + o; o += (size_t)BB * CI * KBASES;  // [B,CI,81]
  float* meanb  = ws + o; o += CI;
  float* istdb  = ws + o; o += CI;

  const dim3 blk(128);
  const float sc = 0.0625f;                      // 256^-0.5
  const int ntN  = (NN + NTILE - 1) / NTILE;     // 77
  const int ntK  = (KBASES + NTILE - 1) / NTILE; // 1
  const int SPLITK = 16;

  // 1) value = w_value @ x + b_value        [CI x N] per batch
  gemm_bf16_wmma<<<dim3(ntN, CI / MT, BB), blk, 0, stream>>>(
      w_value, x, value, b_value, CI, NN, CIN,
      CIN, 1, NN, 1, NN,
      0L, (long)CIN * NN, (long)CI * NN, 1.0f, 1);

  // 2) adaptive avg pool 9x9
  pool9<<<(BB * CI * KBASES + 255) / 256, 256, 0, stream>>>(value, pooled);

  // 3) value_hat = w_vconv @ pooled + b_vconv   [CI x 81]
  gemm_bf16_wmma<<<dim3(ntK, CI / MT, BB), blk, 0, stream>>>(
      w_vconv, pooled, vh, b_vconv, CI, KBASES, CI,
      CI, 1, KBASES, 1, KBASES,
      0L, (long)CI * KBASES, (long)CI * KBASES, 1.0f, 1);

  // 4) sim_bases = scale * value_hat^T @ value   [81 x N]; A[m,k] = vh[k*81+m]
  gemm_bf16_wmma<<<dim3(ntN, (KBASES + MT - 1) / MT, BB), blk, 0, stream>>>(
      vh, value, simb, nullptr, KBASES, NN, CI,
      1, KBASES, NN, 1, NN,
      (long)CI * KBASES, (long)CI * NN, (long)KBASES * NN, sc, 1);

  // 5) softmax over N
  softmax_rows<<<BB * KBASES, 256, 0, stream>>>(simb, NN);

  // 6) bases = value @ sim^T  [CI x 81], K=9801: split-K with f32 atomics
  fill_zero<<<(BB * CI * KBASES + 255) / 256, 256, 0, stream>>>(bases, BB * CI * KBASES);
  gemm_bf16_wmma<<<dim3(ntK, CI / MT, BB * SPLITK), blk, 0, stream>>>(
      value, simb, bases, nullptr, CI, KBASES, NN,
      NN, 1, 1, NN, KBASES,
      (long)CI * NN, (long)KBASES * NN, (long)CI * KBASES, 1.0f, SPLITK);

  // 7) bases += value_hat
  add_inplace<<<(BB * CI * KBASES + 255) / 256, 256, 0, stream>>>(
      bases, vh, BB * CI * KBASES);

  // 8) qraw = w_query @ x + b_query
  gemm_bf16_wmma<<<dim3(ntN, CI / MT, BB), blk, 0, stream>>>(
      w_query, x, qraw, b_query, CI, NN, CIN,
      CIN, 1, NN, 1, NN,
      0L, (long)CIN * NN, (long)CI * NN, 1.0f, 1);

  // 9-10) BatchNorm (training stats over B,N) applied in place
  bn_stats<<<CI, 256, 0, stream>>>(qraw, meanb, istdb);
  {
    size_t total = (size_t)BB * CI * NN;
    bn_apply<<<(unsigned)((total + 255) / 256), 256, 0, stream>>>(
        qraw, meanb, istdb, gamma, beta);
  }

  // 11) sim_map = scale * q^T @ bases      [N x 81]; A[m=n,k=c] = q[c*N+n]
  gemm_bf16_wmma<<<dim3(ntK, (NN + MT - 1) / MT, BB), blk, 0, stream>>>(
      qraw, bases, simmap, nullptr, NN, KBASES, CI,
      1, NN, KBASES, 1, KBASES,
      (long)CI * NN, (long)CI * KBASES, (long)NN * KBASES, sc, 1);

  // 12) softmax over 81 bases
  softmax_rows<<<BB * NN, 256, 0, stream>>>(simmap, KBASES);

  // 13) context = bases @ sim_map^T        [CI x N]; B[k=kb,n] = simmap[n*81+kb]
  gemm_bf16_wmma<<<dim3(ntN, CI / MT, BB), blk, 0, stream>>>(
      bases, simmap, ctx, nullptr, CI, NN, KBASES,
      KBASES, 1, 1, KBASES, NN,
      (long)CI * KBASES, (long)NN * KBASES, (long)CI * NN, 1.0f, 1);

  // 14) out = w_out @ context + b_out      [COUT x N]
  gemm_bf16_wmma<<<dim3(ntN, COUT / MT, BB), blk, 0, stream>>>(
      w_out, ctx, out, b_out, COUT, NN, CI,
      CI, 1, NN, 1, NN,
      0L, (long)CI * NN, (long)COUT * NN, 1.0f, 1);
}
